// Transformer_23055384445158
// MI455X (gfx1250) — compile-verified, matrix operand order checked by reading, not run on hardware
//
#include <hip/hip_runtime.h>
#include <math.h>

typedef __bf16 bf16;
typedef __attribute__((ext_vector_type(16))) __bf16 v16bf;
typedef __attribute__((ext_vector_type(8)))  __bf16 v8bf;
typedef __attribute__((ext_vector_type(8)))  float   v8f;
typedef __attribute__((ext_vector_type(4)))  unsigned int u32x4;
typedef __attribute__((ext_vector_type(4)))  int v4i;

#define DEPTH 6
#define DIM   1024
#define HEADS 16
#define DHEAD 64
#define SEQ   1024
#define BATCH 4
#define ROWS  (BATCH*SEQ)   /* 4096 */
#define FF    4096
#define EPS   1e-5f

// Async global->LDS copy (CDNA5 GLOBAL_LOAD_ASYNC_TO_LDS_B128, ASYNCcnt) if the
// toolchain exposes it; otherwise fall back to VGPR staging (known-good path).
#if __has_builtin(__builtin_amdgcn_global_load_async_to_lds_b128) && \
    __has_builtin(__builtin_amdgcn_s_wait_asynccnt)
#define USE_ASYNC_COPY 1
#else
#define USE_ASYNC_COPY 0
#endif

#if USE_ASYNC_COPY
// Builtin prototype (from the toolchain diagnostic): arg0 = generic v4i*
// (global/flat source), arg1 = addrspace(3) v4i* (LDS dest), then two imms.
typedef __attribute__((address_space(3))) v4i as3v4i;
// generic LDS pointers carry the wave-relative LDS byte offset in their low
// 32 bits (flat aperture rules); as(3) pointers are exactly those 32 bits.
#define GPTR(p) ((v4i*)(size_t)(p))
#define LPTR(p) ((as3v4i*)(unsigned)(size_t)(p))
#endif

// ---------------------------------------------------------------------------
// Fragment loader for 16x16x32 bf16 WMMA (A-frag and B-frag share this layout
// when the B operand is stored "transposed" = each output column's K-vector is
// a contiguous row). Per CDNA5 ISA 7.12.2:
//   lane L holds row r = L&15; lanes 0-15: elems 0-7 -> K 0-7, elems 8-15 -> K16-23
//                              lanes 16-31: elems 0-7 -> K 8-15, elems 8-15 -> K24-31
// base must be 16B aligned and ld (elements) * 2 a multiple of 16.
// ---------------------------------------------------------------------------
__device__ __forceinline__ v16bf frag_ld(const bf16* base, int ld) {
  int lane = threadIdx.x & 31;
  int r    = lane & 15;
  int kb   = (lane < 16) ? 0 : 8;
  const bf16* p = base + r * ld + kb;
  v8bf lo = *(const v8bf*)(p);
  v8bf hi = *(const v8bf*)(p + 16);
  return __builtin_shufflevector(lo, hi, 0,1,2,3,4,5,6,7,8,9,10,11,12,13,14,15);
}

// ---------------------------------------------------------------------------
// GEMM: C[M,N] = A[M,K](bf16,row-major) * Wt[N,K](bf16, pre-transposed weight)
// 256 threads = 8 wave32s; block tile 128x128, K step 32.
// Each wave: 2 (M) x 4 (N) WMMA 16x16 tiles.
// Async path: double-buffered LDS, tile k+1 streams via ASYNCcnt while tile k
// feeds the WMMAs.
// ---------------------------------------------------------------------------
#define BT  128
#define KT  32
#define SLD 40   /* padded LDS row stride (elements): 80B rows, 16B aligned */

template<typename CT>
__global__ __launch_bounds__(256) void k_gemm(const bf16* __restrict__ A,
                                              const bf16* __restrict__ Bt,
                                              CT* __restrict__ C,
                                              int M, int N, int K)
{
  __shared__ __align__(16) bf16 sA[2][BT][SLD];
  __shared__ __align__(16) bf16 sB[2][BT][SLD];

  const int tid  = threadIdx.x;
  const int wid  = tid >> 5;
  const int lane = tid & 31;
  const int wy   = wid & 3;      // 4 waves along M, 32 rows each
  const int wx   = wid >> 2;     // 2 waves along N, 64 cols each
  const int bm   = blockIdx.y * BT;
  const int bn   = blockIdx.x * BT;

  v8f vzero = {0.f,0.f,0.f,0.f,0.f,0.f,0.f,0.f};
  v8f acc[2][4];
#pragma unroll
  for (int i = 0; i < 2; ++i)
#pragma unroll
    for (int j = 0; j < 4; ++j) acc[i][j] = vzero;

#if USE_ASYNC_COPY
  // Each thread owns 2 x 16B chunks per tile (128 rows x 64B / 256 threads).
  auto issue = [&](int buf, int k0) {
#pragma unroll
    for (int it = 0; it < 2; ++it) {
      int item = tid + it * 256;
      int row  = item >> 2;
      int c    = (item & 3) * 8;
      __builtin_amdgcn_global_load_async_to_lds_b128(
          GPTR(A + (size_t)(bm + row) * K + k0 + c),
          LPTR(&sA[buf][row][c]), 0, 0);
      __builtin_amdgcn_global_load_async_to_lds_b128(
          GPTR(Bt + (size_t)(bn + row) * K + k0 + c),
          LPTR(&sB[buf][row][c]), 0, 0);
    }
  };
  issue(0, 0);
  for (int k0 = 0; k0 < K; k0 += KT) {
    const int buf = (k0 / KT) & 1;
    if (k0 + KT < K) {
      issue(buf ^ 1, k0 + KT);               // prefetch next tile into other buf
      __builtin_amdgcn_s_wait_asynccnt(4);   // my 4 loads for tile k are done
    } else {
      __builtin_amdgcn_s_wait_asynccnt(0);
    }
    __syncthreads();                          // everyone's tile-k data landed

    v16bf af[2], bfr[4];
#pragma unroll
    for (int mi = 0; mi < 2; ++mi) af[mi]  = frag_ld(&sA[buf][wy * 32 + mi * 16][0], SLD);
#pragma unroll
    for (int ni = 0; ni < 4; ++ni) bfr[ni] = frag_ld(&sB[buf][wx * 64 + ni * 16][0], SLD);

#pragma unroll
    for (int mi = 0; mi < 2; ++mi)
#pragma unroll
      for (int ni = 0; ni < 4; ++ni)
        acc[mi][ni] = __builtin_amdgcn_wmma_f32_16x16x32_bf16(
            false, af[mi], false, bfr[ni], (short)0, acc[mi][ni], false, false);
    __syncthreads();                          // done reading buf before reuse
  }
#else
  for (int k0 = 0; k0 < K; k0 += KT) {
#pragma unroll
    for (int it = 0; it < 2; ++it) {
      int item = tid + it * 256;
      int row  = item >> 2;
      int c    = (item & 3) * 8;
      *(u32x4*)&sA[0][row][c] = *(const u32x4*)(A  + (size_t)(bm + row) * K + k0 + c);
      *(u32x4*)&sB[0][row][c] = *(const u32x4*)(Bt + (size_t)(bn + row) * K + k0 + c);
    }
    if (k0 + KT < K) { // hint the next K tile into cache (global_prefetch_b8)
      __builtin_prefetch(A  + (size_t)(bm + (tid >> 1)) * K + k0 + KT, 0, 1);
      __builtin_prefetch(Bt + (size_t)(bn + (tid >> 1)) * K + k0 + KT, 0, 1);
    }
    __syncthreads();

    v16bf af[2], bfr[4];
#pragma unroll
    for (int mi = 0; mi < 2; ++mi) af[mi]  = frag_ld(&sA[0][wy * 32 + mi * 16][0], SLD);
#pragma unroll
    for (int ni = 0; ni < 4; ++ni) bfr[ni] = frag_ld(&sB[0][wx * 64 + ni * 16][0], SLD);

#pragma unroll
    for (int mi = 0; mi < 2; ++mi)
#pragma unroll
      for (int ni = 0; ni < 4; ++ni)
        acc[mi][ni] = __builtin_amdgcn_wmma_f32_16x16x32_bf16(
            false, af[mi], false, bfr[ni], (short)0, acc[mi][ni], false, false);
    __syncthreads();
  }
#endif

  // C layout: VGPR r -> row r (lanes 0-15) / r+8 (lanes 16-31); col = lane&15
  const int half = lane >> 4;
  const int n    = lane & 15;
#pragma unroll
  for (int mi = 0; mi < 2; ++mi)
#pragma unroll
    for (int ni = 0; ni < 4; ++ni)
#pragma unroll
      for (int r = 0; r < 8; ++r) {
        int row = bm + wy * 32 + mi * 16 + r + half * 8;
        int col = bn + wx * 64 + ni * 16 + n;
        C[(size_t)row * N + col] = (CT)acc[mi][ni][r];
      }
}

// ---------------------------------------------------------------------------
// Weight convert + transpose: W[K,N] fp32 -> Wt[N,K] bf16
// ---------------------------------------------------------------------------
__global__ void k_wt(const float* __restrict__ W, bf16* __restrict__ Wt, int K, int N)
{
  size_t idx = (size_t)blockIdx.x * 256 + threadIdx.x;
  size_t total = (size_t)K * N;
  if (idx >= total) return;
  int k = (int)(idx / N);
  int n = (int)(idx % N);
  Wt[(size_t)n * K + k] = (bf16)W[idx];
}

// ---------------------------------------------------------------------------
// LayerNorm (biased variance, gain, no bias). One block per row.
// ---------------------------------------------------------------------------
template<typename OutT>
__global__ __launch_bounds__(256) void k_ln(const float* __restrict__ in,
                                            const float* __restrict__ g,
                                            OutT* __restrict__ out, int D)
{
  __shared__ float rs[8], rss[8];
  __shared__ float mean_s, rstd_s;
  int row = blockIdx.x;
  const float* x = in + (size_t)row * D;
  float s = 0.f, ss = 0.f;
  for (int i = threadIdx.x; i < D; i += 256) { float v = x[i]; s += v; ss += v * v; }
  for (int off = 16; off > 0; off >>= 1) { s += __shfl_xor(s, off); ss += __shfl_xor(ss, off); }
  int wid = threadIdx.x >> 5;
  if ((threadIdx.x & 31) == 0) { rs[wid] = s; rss[wid] = ss; }
  __syncthreads();
  if (threadIdx.x == 0) {
    float ts = 0.f, tss = 0.f;
    for (int w = 0; w < 8; ++w) { ts += rs[w]; tss += rss[w]; }
    float mean = ts / D;
    float var  = tss / D - mean * mean;
    mean_s = mean;
    rstd_s = rsqrtf(var + EPS);
  }
  __syncthreads();
  float mean = mean_s, rstd = rstd_s;
  OutT* o = out + (size_t)row * D;
  for (int i = threadIdx.x; i < D; i += 256)
    o[i] = (OutT)((x[i] - mean) * rstd * g[i]);
}

// LayerNorm then residual add:  res += LN(in)*g
__global__ __launch_bounds__(256) void k_ln_add(const float* __restrict__ in,
                                                const float* __restrict__ g,
                                                float* __restrict__ res, int D)
{
  __shared__ float rs[8], rss[8];
  __shared__ float mean_s, rstd_s;
  int row = blockIdx.x;
  const float* x = in + (size_t)row * D;
  float s = 0.f, ss = 0.f;
  for (int i = threadIdx.x; i < D; i += 256) { float v = x[i]; s += v; ss += v * v; }
  for (int off = 16; off > 0; off >>= 1) { s += __shfl_xor(s, off); ss += __shfl_xor(ss, off); }
  int wid = threadIdx.x >> 5;
  if ((threadIdx.x & 31) == 0) { rs[wid] = s; rss[wid] = ss; }
  __syncthreads();
  if (threadIdx.x == 0) {
    float ts = 0.f, tss = 0.f;
    for (int w = 0; w < 8; ++w) { ts += rs[w]; tss += rss[w]; }
    float mean = ts / D;
    float var  = tss / D - mean * mean;
    mean_s = mean;
    rstd_s = rsqrtf(var + EPS);
  }
  __syncthreads();
  float mean = mean_s, rstd = rstd_s;
  float* r = res + (size_t)row * D;
  for (int i = threadIdx.x; i < D; i += 256)
    r[i] += (x[i] - mean) * rstd * g[i];
}

// GEGLU: h[ROWS][2*FF] bf16 -> o[ROWS][FF] fp32 ; o = a * gelu_exact(gate)
__global__ void k_geglu(const bf16* __restrict__ h, float* __restrict__ o)
{
  size_t idx = (size_t)blockIdx.x * 256 + threadIdx.x;
  size_t row = idx >> 12;       // / 4096
  size_t col = idx & 4095;
  float a  = (float)h[row * (2 * FF) + col];
  float gt = (float)h[row * (2 * FF) + FF + col];
  float gl = 0.5f * gt * (1.f + erff(gt * 0.70710678118654752f));
  o[idx] = a * gl;
}

__global__ void k_add(float* __restrict__ res, const float* __restrict__ v, int total)
{
  size_t i = (size_t)blockIdx.x * 256 + threadIdx.x;
  if (i < (size_t)total) res[i] += v[i];
}

// ---------------------------------------------------------------------------
// Flash attention (causal, 16 heads, d=64) with bf16 WMMA.
// Grid: BATCH*HEADS*(SEQ/64) blocks of 128 threads (4 wave32s).
// Each wave owns 16 q rows; block shares K/V tiles (32 keys) in LDS.
// qkv: [B*SEQ, 3*DIM] fp32 (q | k | v sections). out: bf16 [B*SEQ, DIM].
// ---------------------------------------------------------------------------
#define SK_LD 72
#define SV_LD 40
#define SP_LD 40

__global__ __launch_bounds__(128) void k_attn(const float* __restrict__ qkv,
                                              bf16* __restrict__ out)
{
  __shared__ __align__(16) bf16 sK[32][SK_LD];        // key rows [key][d]
  __shared__ __align__(16) bf16 sV[DHEAD][SV_LD];     // V transposed [d][key]
  __shared__ __align__(16) bf16 sP[4][16][SP_LD];     // per-wave P tile [m][key]

  const int tid  = threadIdx.x;
  const int wid  = tid >> 5;
  const int lane = tid & 31;
  const int half = lane >> 4;
  const int n    = lane & 15;

  int bid = blockIdx.x;
  int qt  = bid & 15;
  int h   = (bid >> 4) & 15;
  int b   = bid >> 8;
  int q0    = qt * 64;
  int qrow0 = q0 + wid * 16;
  const float scale = 0.125f;   // 64^-0.5

  const float* base = qkv + (size_t)b * SEQ * (3 * DIM);

  // Q fragments (2 K-steps of 32 over d=64), scale folded in
  v16bf qf[2];
  {
    int kb = half * 8;
    const float* qp = base + (size_t)(qrow0 + n) * (3 * DIM) + h * DHEAD;
#pragma unroll
    for (int f = 0; f < 2; ++f) {
      v16bf t;
#pragma unroll
      for (int i = 0; i < 8; ++i) {
        t[i]     = (bf16)(qp[f * 32 + kb + i] * scale);
        t[i + 8] = (bf16)(qp[f * 32 + kb + 16 + i] * scale);
      }
      qf[f] = t;
    }
  }

  v8f vzero = {0.f,0.f,0.f,0.f,0.f,0.f,0.f,0.f};
  float rmax[8], rsum[8];
  v8f   oacc[4];
#pragma unroll
  for (int r = 0; r < 8; ++r) { rmax[r] = -3.0e38f; rsum[r] = 0.f; }
#pragma unroll
  for (int d = 0; d < 4; ++d) oacc[d] = vzero;

  const int nkt = (qt + 1) * 2;                 // causal bound, 32-key tiles
  for (int kt = 0; kt < nkt; ++kt) {
    int j0 = kt * 32;
    { // stage K tile (rows) and V tile (transposed) as bf16
      int key = tid >> 2;
      int seg = (tid & 3) * 16;
      const float* kp = base + (size_t)(j0 + key) * (3 * DIM) + DIM     + h * DHEAD + seg;
      const float* vp = base + (size_t)(j0 + key) * (3 * DIM) + 2 * DIM + h * DHEAD + seg;
#pragma unroll
      for (int i = 0; i < 16; ++i) {
        sK[key][seg + i] = (bf16)kp[i];
        sV[seg + i][key] = (bf16)vp[i];
      }
    }
    __syncthreads();

    // S = Q @ K^T : two 16-key column tiles, 2 K-steps each
    v8f s[2]; s[0] = vzero; s[1] = vzero;
#pragma unroll
    for (int ni = 0; ni < 2; ++ni)
#pragma unroll
      for (int f = 0; f < 2; ++f) {
        v16bf kf = frag_ld(&sK[ni * 16][f * 32], SK_LD);
        s[ni] = __builtin_amdgcn_wmma_f32_16x16x32_bf16(
            false, qf[f], false, kf, (short)0, s[ni], false, false);
      }

    // causal mask + online softmax (rows live in 16-lane halves)
#pragma unroll
    for (int r = 0; r < 8; ++r) {
      int qi = qrow0 + r + half * 8;
#pragma unroll
      for (int ni = 0; ni < 2; ++ni) {
        int j = j0 + ni * 16 + n;
        if (j > qi) s[ni][r] = -1.0e30f;
      }
      float mx = fmaxf(s[0][r], s[1][r]);
#pragma unroll
      for (int off = 1; off < 16; off <<= 1) mx = fmaxf(mx, __shfl_xor(mx, off));
      float nm    = fmaxf(rmax[r], mx);
      float alpha = expf(rmax[r] - nm);
      float e0 = expf(s[0][r] - nm);
      float e1 = expf(s[1][r] - nm);
      float ls = e0 + e1;
#pragma unroll
      for (int off = 1; off < 16; off <<= 1) ls += __shfl_xor(ls, off);
      rsum[r] = rsum[r] * alpha + ls;
      rmax[r] = nm;
#pragma unroll
      for (int d = 0; d < 4; ++d) oacc[d][r] *= alpha;
      int m = r + half * 8;
      sP[wid][m][n]      = (bf16)e0;   // C-layout -> A-layout via per-wave LDS
      sP[wid][m][16 + n] = (bf16)e1;   // (DS ops are in-order within a wave)
    }

    // O += P @ V  (K = 32 keys, 4 d-tiles of 16)
    v16bf pf = frag_ld(&sP[wid][0][0], SP_LD);
#pragma unroll
    for (int d = 0; d < 4; ++d) {
      v16bf vf = frag_ld(&sV[d * 16][0], SV_LD);
      oacc[d] = __builtin_amdgcn_wmma_f32_16x16x32_bf16(
          false, pf, false, vf, (short)0, oacc[d], false, false);
    }
    __syncthreads();
  }

  // normalize and emit bf16 (feeds the out-projection GEMM directly)
#pragma unroll
  for (int d = 0; d < 4; ++d)
#pragma unroll
    for (int r = 0; r < 8; ++r) {
      int row = b * SEQ + qrow0 + r + half * 8;
      int col = h * DHEAD + d * 16 + n;
      out[(size_t)row * DIM + col] = (bf16)(oacc[d][r] / rsum[r]);
    }
}

// ---------------------------------------------------------------------------
// Orchestration
// ---------------------------------------------------------------------------
extern "C" void kernel_launch(void* const* d_in, const int* in_sizes, int n_in,
                              void* d_out, int out_size, void* d_ws, size_t ws_size,
                              hipStream_t stream)
{
  const float* x_in    = (const float*)d_in[0];
  const float* qkv_w   = (const float*)d_in[1];
  const float* out_w   = (const float*)d_in[2];
  const float* ff_w1   = (const float*)d_in[3];
  const float* ff_w2   = (const float*)d_in[4];
  const float* attn_g  = (const float*)d_in[5];
  const float* outln_g = (const float*)d_in[6];
  const float* ffn_g   = (const float*)d_in[7];
  const float* ffln_g  = (const float*)d_in[8];
  const float* nin_g   = (const float*)d_in[9];
  const float* nout_g  = (const float*)d_in[10];

  char* w = (char*)d_ws;
  size_t off = 0;
  auto alloc = [&](size_t bytes) {
    char* p = w + off;
    off = (off + bytes + 255) & ~(size_t)255;
    return p;
  };
  float* xres = (float*)alloc((size_t)ROWS * DIM * 4);        // residual stream
  float* qkvb = (float*)alloc((size_t)ROWS * 3 * DIM * 4);    // qkv GEMM out
  bf16*  hbuf = (bf16*) alloc((size_t)ROWS * 2 * FF * 2);     // ff1 out (bf16)
  float* gbuf = (float*)alloc((size_t)ROWS * FF * 4);         // generic fp32
  bf16*  abuf = (bf16*) alloc((size_t)ROWS * FF * 2);         // GEMM A (bf16)
  bf16*  wtbf = (bf16*) alloc((size_t)DIM * 2 * FF * 2);      // transposed W

  dim3 blk(256);

  // x = LN(x, norm_in_g)
  k_ln<float><<<ROWS, blk, 0, stream>>>(x_in, nin_g, xres, DIM);

  for (int l = 0; l < DEPTH; ++l) {
    const float* qw = qkv_w + (size_t)l * DIM * 3 * DIM;
    const float* ow = out_w + (size_t)l * DIM * DIM;
    const float* f1 = ff_w1 + (size_t)l * DIM * 2 * FF;
    const float* f2 = ff_w2 + (size_t)l * FF * DIM;

    // ---- attention block ----
    k_ln<bf16><<<ROWS, blk, 0, stream>>>(xres, attn_g + (size_t)l * DIM, abuf, DIM);
    { size_t t = (size_t)DIM * 3 * DIM;
      k_wt<<<(unsigned)((t + 255) / 256), blk, 0, stream>>>(qw, wtbf, DIM, 3 * DIM); }
    k_gemm<float><<<dim3(3 * DIM / 128, ROWS / 128), blk, 0, stream>>>(
        abuf, wtbf, qkvb, ROWS, 3 * DIM, DIM);
    k_attn<<<BATCH * HEADS * (SEQ / 64), dim3(128), 0, stream>>>(qkvb, abuf);
    { size_t t = (size_t)DIM * DIM;
      k_wt<<<(unsigned)((t + 255) / 256), blk, 0, stream>>>(ow, wtbf, DIM, DIM); }
    k_gemm<float><<<dim3(DIM / 128, ROWS / 128), blk, 0, stream>>>(
        abuf, wtbf, gbuf, ROWS, DIM, DIM);
    k_ln_add<<<ROWS, blk, 0, stream>>>(gbuf, outln_g + (size_t)l * DIM, xres, DIM);

    // ---- GEGLU feedforward block ----
    k_ln<bf16><<<ROWS, blk, 0, stream>>>(xres, ffn_g + (size_t)l * DIM, abuf, DIM);
    { size_t t = (size_t)DIM * 2 * FF;
      k_wt<<<(unsigned)((t + 255) / 256), blk, 0, stream>>>(f1, wtbf, DIM, 2 * FF); }
    k_gemm<bf16><<<dim3(2 * FF / 128, ROWS / 128), blk, 0, stream>>>(
        abuf, wtbf, hbuf, ROWS, 2 * FF, DIM);
    { size_t t = (size_t)ROWS * FF;
      k_geglu<<<(unsigned)(t / 256), blk, 0, stream>>>(hbuf, gbuf); }
    k_ln<bf16><<<ROWS, blk, 0, stream>>>(gbuf, ffln_g + (size_t)l * FF, abuf, FF);
    { size_t t = (size_t)FF * DIM;
      k_wt<<<(unsigned)((t + 255) / 256), blk, 0, stream>>>(f2, wtbf, FF, DIM); }
    k_gemm<float><<<dim3(DIM / 128, ROWS / 128), blk, 0, stream>>>(
        abuf, wtbf, gbuf, ROWS, DIM, FF);
    k_add<<<(ROWS * DIM) / 256, blk, 0, stream>>>(xres, gbuf, ROWS * DIM);
  }

  // out = LN(x, norm_out_g)
  k_ln<float><<<ROWS, blk, 0, stream>>>(xres, nout_g, (float*)d_out, DIM);
}
